// Agent_3968549781916
// MI455X (gfx1250) — matrix operand order repaired and checked by reading, not compiled, not run on hardware
//
#include <hip/hip_runtime.h>
#include <hip/hip_bf16.h>

typedef __attribute__((ext_vector_type(16))) _Float16 v16h;
typedef __attribute__((ext_vector_type(8)))  float    v8f;

static __device__ __forceinline__ v8f wmma_f16(v16h a, v16h b, v8f c) {
  // D = A(16x32 f16) * B(32x16 f16) + C(16x16 f32)
  return __builtin_amdgcn_wmma_f32_16x16x32_f16(false, a, false, b, (short)0, c, false, false);
}

// ---------------------------------------------------------------------------
// One-shot f32 -> f16 weight conversion (removes v_cvt from GEMM hot loops).
// ---------------------------------------------------------------------------
__global__ __launch_bounds__(256)
void cvt_w_k(const float* __restrict__ src, _Float16* __restrict__ dst, int n) {
  const int i = blockIdx.x * blockDim.x + threadIdx.x;
  if (i < n) dst[i] = (_Float16)src[i];
}

// ---------------------------------------------------------------------------
// Implicit-im2col conv as WMMA GEMM, TN N-tiles per wave (TN*16 == Cout, so
// each wave produces all output channels for its 16 pixels -> A loaded once).
//   M = Nimg*Hout*Wout, N = Cout, K = Cin*KH*KW (all divide exactly).
// ---------------------------------------------------------------------------
template<int Cin,int Hin,int Win,int KH,int KW,int STR,
         int Cout,int Hout,int Wout,int TN,bool IN_F32>
__global__ __launch_bounds__(256)
void conv_wmma_k(const void* __restrict__ inp, const _Float16* __restrict__ w16,
                 const float* __restrict__ bias, _Float16* __restrict__ out,
                 int Nimg) {
  constexpr int K  = Cin * KH * KW;
  constexpr int HW = Hout * Wout;
  static_assert(TN * 16 == Cout, "wave covers all channels");
  const int  lane = threadIdx.x & 31;
  const int  half = lane >> 4;
  const int  r    = lane & 15;
  const long mt   = (long)blockIdx.x * 8 + (threadIdx.x >> 5);
  const long MT   = (long)Nimg * HW / 16;
  if (mt >= MT) return;               // wave-uniform

  // Per-lane A row: lane holds row M = m0 + r (both lane halves)
  const long M  = mt * 16 + r;
  const int  n  = (int)(M / HW);
  const int  rm = (int)(M % HW);
  const int  oy = rm / Wout, ox = rm % Wout;
  const long ibase = (long)n * Cin * Hin * Win + (long)(oy * STR) * Win + (ox * STR);

  v8f acc[TN];
#pragma unroll
  for (int t = 0; t < TN; ++t) acc[t] = v8f{};

  for (int k0 = 0; k0 < K; k0 += 32) {
    v16h a;
#pragma unroll
    for (int v = 0; v < 8; ++v) {
#pragma unroll
      for (int e = 0; e < 2; ++e) {
        // 16-bit A 16x32 layout: VGPR slot -> K index
        const int kk = k0 + (v >> 2) * 16 + half * 8 + (v & 3) * 2 + e;
        const int c  = kk / (KH * KW);
        const int kr = kk % (KH * KW);
        const int ky = kr / KW, kx = kr % KW;
        const long ad = ibase + (long)c * (Hin * Win) + (long)ky * Win + kx;
        float x;
        if constexpr (IN_F32) x = ((const float*)inp)[ad] * (1.0f / 255.0f);
        else                  x = (float)((const _Float16*)inp)[ad];
        a[v * 2 + e] = (_Float16)x;
      }
    }
#pragma unroll
    for (int t = 0; t < TN; ++t) {
      const _Float16* wrow = w16 + (long)(t * 16 + r) * K;  // B col = channel
      v16h b;
#pragma unroll
      for (int j = 0; j < 16; ++j)    // B 32x16: lane group h holds K=k0+(j>>3)*16+8h+(j&7)
        b[j] = wrow[k0 + (j >> 3) * 16 + half * 8 + (j & 7)];
      acc[t] = wmma_f16(a, b, acc[t]);
    }
  }

#pragma unroll
  for (int i = 0; i < 8; ++i) {       // C layout: VGPR i -> row m0 + 8*half + i
    const long Mo = mt * 16 + half * 8 + i;
    const int  no = (int)(Mo / HW);
    const int  ro = (int)(Mo % HW);
#pragma unroll
    for (int t = 0; t < TN; ++t) {
      const int col = t * 16 + r;
      float v = acc[t][i] + bias[col];
      v = v > 0.f ? v : 0.f;          // ReLU
      out[((long)no * Cout + col) * HW + ro] = (_Float16)v;
    }
  }
}

// ---------------------------------------------------------------------------
// Dense linear, TN N-tiles per wave: C = relu?(A[M,K](f16) * W16[N,K]^T + b)
// ---------------------------------------------------------------------------
template<int K, int Nout, int TN, bool RELU, bool OUT16>
__global__ __launch_bounds__(256)
void linear_wmma_k(const _Float16* __restrict__ A, const _Float16* __restrict__ W16,
                   const float* __restrict__ bias, void* __restrict__ out, int Mrows) {
  constexpr int NG = Nout / (16 * TN);        // N-tile groups
  const int  lane = threadIdx.x & 31;
  const int  half = lane >> 4, r = lane & 15;
  const long wid  = (long)blockIdx.x * 8 + (threadIdx.x >> 5);
  if (wid >= (long)(Mrows / 16) * NG) return;
  const long mt = wid / NG;
  const int  ng = (int)(wid % NG);
  const _Float16* arow = A + (mt * 16 + r) * (long)K;

  v8f acc[TN];
#pragma unroll
  for (int t = 0; t < TN; ++t) acc[t] = v8f{};

  for (int k0 = 0; k0 < K; k0 += 32) {
    v16h a;
#pragma unroll
    for (int v = 0; v < 8; ++v)
#pragma unroll
      for (int e = 0; e < 2; ++e) {
        const int kk = k0 + (v >> 2) * 16 + half * 8 + (v & 3) * 2 + e;
        a[v * 2 + e] = arow[kk];
      }
#pragma unroll
    for (int t = 0; t < TN; ++t) {
      const int col = (ng * TN + t) * 16 + r;
      const _Float16* wrow = W16 + (long)col * K;
      v16h b;
#pragma unroll
      for (int j = 0; j < 16; ++j)
        b[j] = wrow[k0 + (j >> 3) * 16 + half * 8 + (j & 7)];
      acc[t] = wmma_f16(a, b, acc[t]);
    }
  }
#pragma unroll
  for (int i = 0; i < 8; ++i) {
    const long Mo = mt * 16 + half * 8 + i;
#pragma unroll
    for (int t = 0; t < TN; ++t) {
      const int col = (ng * TN + t) * 16 + r;
      float v = acc[t][i] + bias[col];
      if (RELU) v = v > 0.f ? v : 0.f;
      if (OUT16) ((_Float16*)out)[Mo * Nout + col] = (_Float16)v;
      else       ((float*)out)[Mo * Nout + col]    = v;
    }
  }
}

// ---------------------------------------------------------------------------
// Persistent GRU scan. One workgroup, 24 waves (768 threads).
// h (f32, for gates) + h16 (f16 mirror, for WMMA A-frags) + gh live in LDS.
// w_hh B-fragments preloaded to VGPRs once, reused for all 128 steps.
// ---------------------------------------------------------------------------
__global__ __launch_bounds__(768)
void gru_k(const float* __restrict__ gx, const float* __restrict__ done,
           const float* __restrict__ h0, const float* __restrict__ w_hh,
           const float* __restrict__ b_hh, float* __restrict__ hs,
           float* __restrict__ hfin) {
  __shared__ float    h_lds[32 * 128];   // 16 KB (f32 masked h, for gate math)
  __shared__ _Float16 h16 [32 * 128];    //  8 KB (f16 mirror for WMMA)
  __shared__ float    gh_lds[32 * 384];  // 48 KB
  const int tid  = threadIdx.x;
  const int lane = tid & 31, wave = tid >> 5;   // wave 0..23
  const int half = lane >> 4, r = lane & 15;
  const int col  = wave * 16 + r;               // gate index g in [0,384)

  // Preload w_hh B-fragments (K=128 -> 4 fragments of 32)
  v16h bfrag[4];
#pragma unroll
  for (int ks = 0; ks < 4; ++ks)
#pragma unroll
    for (int j = 0; j < 16; ++j)
      bfrag[ks][j] = (_Float16)w_hh[(long)col * 128 + ks * 32 + (j >> 3) * 16 + half * 8 + (j & 7)];
  const float bhh = b_hh[col];

  for (int idx = tid; idx < 4096; idx += 768) h_lds[idx] = h0[idx];
  __syncthreads();

  for (int t = 0; t < 128; ++t) {
    // done-reset mask: hprev *= (1 - d[t,b]); refresh f16 mirror
    for (int idx = tid; idx < 4096; idx += 768) {
      const float hv = h_lds[idx] * (1.0f - done[t * 32 + (idx >> 7)]);
      h_lds[idx] = hv;
      h16[idx]   = (_Float16)hv;
    }
    __syncthreads();

    // gh = h @ w_hh^T + b_hh  (M=32 -> two 16-row tiles per wave)
#pragma unroll
    for (int mtile = 0; mtile < 2; ++mtile) {
      v8f acc = {};
#pragma unroll
      for (int ks = 0; ks < 4; ++ks) {
        v16h a;
#pragma unroll
        for (int v = 0; v < 8; ++v)
#pragma unroll
          for (int e = 0; e < 2; ++e) {
            const int kk = ks * 32 + (v >> 2) * 16 + half * 8 + (v & 3) * 2 + e;
            a[v * 2 + e] = h16[(mtile * 16 + r) * 128 + kk];
          }
        acc = wmma_f16(a, bfrag[ks], acc);
      }
#pragma unroll
      for (int i = 0; i < 8; ++i) {
        const int brow = mtile * 16 + half * 8 + i;
        gh_lds[brow * 384 + col] = acc[i] + bhh;
      }
    }
    __syncthreads();

    // gates (f32)
    float hn[6];
#pragma unroll
    for (int it = 0; it < 6; ++it) {
      const int idx = tid + it * 768;
      if (idx < 4096) {
        const int b = idx >> 7, j = idx & 127;
        const float* g  = gx + ((long)t * 32 + b) * 384;
        const float* gh = gh_lds + b * 384;
        const float rr = 1.0f / (1.0f + __expf(-(g[j]       + gh[j])));
        const float zz = 1.0f / (1.0f + __expf(-(g[128 + j] + gh[128 + j])));
        const float nn = tanhf(g[256 + j] + rr * gh[256 + j]);
        hn[it] = (1.0f - zz) * nn + zz * h_lds[idx];
      }
    }
    __syncthreads();
#pragma unroll
    for (int it = 0; it < 6; ++it) {
      const int idx = tid + it * 768;
      if (idx < 4096) {
        h_lds[idx] = hn[it];
        hs[(long)t * 4096 + idx] = hn[it];
      }
    }
    __syncthreads();
  }
  for (int idx = tid; idx < 4096; idx += 768) hfin[idx] = h_lds[idx];
}

// ---------------------------------------------------------------------------
// Actor/critic heads: 7 dot products of length 128 per row (tiny -> VALU).
// ---------------------------------------------------------------------------
__global__ __launch_bounds__(256)
void heads_k(const float* __restrict__ hs, const float* __restrict__ aw,
             const float* __restrict__ ab, const float* __restrict__ cw,
             const float* __restrict__ cb, float* __restrict__ out) {
  const int idx = blockIdx.x * blockDim.x + threadIdx.x;
  if (idx >= 4096 * 7) return;
  const int row = idx / 7, c = idx % 7;
  const float* h = hs + (long)row * 128;
  const float* w = (c < 6) ? (aw + c * 128) : cw;
  float s = (c < 6) ? ab[c] : cb[0];
#pragma unroll 8
  for (int k = 0; k < 128; ++k) s += h[k] * w[k];
  if (c < 6) out[(long)row * 6 + c] = s;
  else       out[24576 + row]       = s;
}

extern "C" void kernel_launch(void* const* d_in, const int* in_sizes, int n_in,
                              void* d_out, int out_size, void* d_ws, size_t ws_size,
                              hipStream_t stream) {
  // inputs per setup_inputs() order
  const float* x    = (const float*)d_in[0];   // [4096,1,84,84]
  const float* done = (const float*)d_in[1];   // [4096]
  const float* h0   = (const float*)d_in[2];   // [1,32,128]
  const float* c1w = (const float*)d_in[3];  const float* c1b = (const float*)d_in[4];
  const float* c2w = (const float*)d_in[5];  const float* c2b = (const float*)d_in[6];
  const float* c3w = (const float*)d_in[7];  const float* c3b = (const float*)d_in[8];
  const float* fcw = (const float*)d_in[9];  const float* fcb = (const float*)d_in[10];
  const float* wih = (const float*)d_in[11]; const float* whh = (const float*)d_in[12];
  const float* bih = (const float*)d_in[13]; const float* bhh = (const float*)d_in[14];
  const float* aw  = (const float*)d_in[15]; const float* ab  = (const float*)d_in[16];
  const float* cw  = (const float*)d_in[17]; const float* cb  = (const float*)d_in[18];
  float* out = (float*)d_out;   // [logits 4096*6 | value 4096 | h_final 4096]

  // workspace layout (a3/feat/gx/hs alias dead a1 region; peak ~151 MB)
  char* ws = (char*)d_ws;
  _Float16* a1   = (_Float16*)(ws);                 // 4096*32*400 f16 = 104,857,600 B
  _Float16* a2   = (_Float16*)(ws + 104857600);     // 4096*64*81  f16 =  42,467,328 B
  _Float16* a3   = (_Float16*)(ws);                 // 4096*64*49  f16 (aliases a1)
  _Float16* feat = (_Float16*)(ws + 25690112);      // 4096*512 f16
  float*    gx   = (float*)   (ws + 29884416);      // 4096*384 f32
  float*    hs   = (float*)   (ws + 36175872);      // 4096*128 f32
  // preconverted f16 weights (after a2)
  _Float16* w1h  = (_Float16*)(ws + 147324928);     //  32*64
  _Float16* w2h  = (_Float16*)(ws + 147329024);     //  64*512
  _Float16* w3h  = (_Float16*)(ws + 147394560);     //  64*576
  _Float16* fch  = (_Float16*)(ws + 147468288);     // 512*3136
  _Float16* wihh = (_Float16*)(ws + 150679552);     // 384*512
  const int Nimg = 4096;

  // 0) weight f32 -> f16 (one-shot, tiny)
  cvt_w_k<<<8,    256, 0, stream>>>(c1w, w1h, 32 * 64);
  cvt_w_k<<<128,  256, 0, stream>>>(c2w, w2h, 64 * 512);
  cvt_w_k<<<144,  256, 0, stream>>>(c3w, w3h, 64 * 576);
  cvt_w_k<<<6272, 256, 0, stream>>>(fcw, fch, 512 * 3136);
  cvt_w_k<<<768,  256, 0, stream>>>(wih, wihh, 384 * 512);

  // conv1: [4096,1,84,84] -> [4096,32,20,20]   (M=1,638,400 N=32 K=64)
  conv_wmma_k<1, 84, 84, 8, 8, 4, 32, 20, 20, 2, true>
      <<<12800, 256, 0, stream>>>((const void*)x, w1h, c1b, a1, Nimg);
  // conv2: -> [4096,64,9,9]                    (M=331,776 N=64 K=512)
  conv_wmma_k<32, 20, 20, 4, 4, 2, 64, 9, 9, 4, false>
      <<<2592, 256, 0, stream>>>((const void*)a1, w2h, c2b, a2, Nimg);
  // conv3: -> [4096,64,7,7]                    (M=200,704 N=64 K=576)
  conv_wmma_k<64, 9, 9, 3, 3, 1, 64, 7, 7, 4, false>
      <<<1568, 256, 0, stream>>>((const void*)a2, w3h, c3b, a3, Nimg);
  // fc: [4096,3136] -> [4096,512] f16, ReLU    (TN=4 -> 8 N-groups)
  linear_wmma_k<3136, 512, 4, true, true>
      <<<256, 256, 0, stream>>>(a3, fch, fcb, (void*)feat, Nimg);
  // gx: [4096,512] -> [4096,384] f32 (+b_ih)   (TN=4 -> 6 N-groups)
  linear_wmma_k<512, 384, 4, false, false>
      <<<192, 256, 0, stream>>>(feat, wihh, bih, (void*)gx, Nimg);
  // GRU scan: persistent single workgroup, writes hs and h_final
  gru_k<<<1, 768, 0, stream>>>(gx, done, h0, whh, bhh, hs, out + 28672);
  // heads: logits + value
  heads_k<<<112, 256, 0, stream>>>(hs, aw, ab, cw, cb, out);
}
